// SIM_66194035966220
// MI455X (gfx1250) — compile-verified, matrix operand order checked by reading, not run on hardware
//
#include <hip/hip_runtime.h>
#include <stdint.h>

// ---------------------------------------------------------------------------
// DIN forward for MI455X (gfx1250, wave32, WMMA).
// H=64, B=4096, L=200. Activation unit folded:
//   [h,it,h-it,h*it]@W1 = h@(W0+W2) + it@(W1-W2) + (h*it)@W3
// Hot GEMM done with v_wmma_f32_16x16x32_f16 (M=16 hist rows, N=48 pad, K=128).
// ---------------------------------------------------------------------------

#define Hh   64
#define Bb   4096
#define Ll   200
#define NTILES_L 13          // ceil(200/16)
#define NPAD 48              // 36 -> 3 x 16 N-tiles
#define THRE 0.05f
#define EPSR 1e-8f

typedef __attribute__((ext_vector_type(16))) _Float16 v16h;
typedef __attribute__((ext_vector_type(8)))  float    v8f;

union Frag { v16h v; uint4 q[2]; };

__device__ __forceinline__ void wave_ds_sync() {
  // DS ops of one wave complete in order; wait so this wave's LDS stores are
  // visible to its own subsequent ds loads (CDNA5 split counters).
  asm volatile("s_wait_dscnt 0" ::: "memory");
}

// ---------------------------------------------------------------------------
// Kernel 0: fold au_w1 and pack WMMA B-fragments.
//   Wh = W[0:128]+W[256:384], Wp = W[384:512]  -> f16, per-lane packed
//   Wit = W[128:256]-W[256:384]                -> f32 [128][36]
// B-fragment layout (16-bit, 32x16, mirrors ISA A table):
//   lane l: n = l&15 ; VGPR0-3: K = kb+0..7, VGPR4-7: K = kb+16..23, kb=(l>>4)*8
// Packed f16 index: ((nt*4+kc)*32 + lane)*16 + j
// ---------------------------------------------------------------------------
__global__ void din_prep(const float* __restrict__ au_w1,
                         float* __restrict__ Wit,
                         _Float16* __restrict__ Bh,
                         _Float16* __restrict__ Bp) {
  int idx = blockIdx.x * blockDim.x + threadIdx.x;
  if (idx < 128 * 36) {
    int k = idx / 36, n = idx % 36;
    Wit[idx] = au_w1[(128 + k) * 36 + n] - au_w1[(256 + k) * 36 + n];
  }
  if (idx < 6144) {
    int j    = idx & 15;
    int lane = (idx >> 4) & 31;
    int kc   = (idx >> 9) & 3;
    int nt   = idx >> 11;                       // 0..2
    int n    = nt * 16 + (lane & 15);
    int kk   = kc * 32 + ((j >> 3) << 4) + ((lane >> 4) << 3) + (j & 7);
    float vh = 0.f, vp = 0.f;
    if (n < 36) {
      vh = au_w1[kk * 36 + n] + au_w1[(256 + kk) * 36 + n];
      vp = au_w1[(384 + kk) * 36 + n];
    }
    Bh[idx] = (_Float16)vh;
    Bp[idx] = (_Float16)vp;
  }
}

// ---------------------------------------------------------------------------
// Kernel 1: attention. One 128-thread block (4 waves) per batch row.
// ---------------------------------------------------------------------------
__global__ void __launch_bounds__(128)
din_attn(const int* __restrict__ user, const int* __restrict__ item,
         const int* __restrict__ cate, const int* __restrict__ timei,
         const int* __restrict__ hist,
         const float* __restrict__ user_emb, const float* __restrict__ item_emb,
         const float* __restrict__ cate_emb, const float* __restrict__ time_emb,
         const float* __restrict__ au_b1, const float* __restrict__ au_w2,
         const float* __restrict__ au_b2,
         const float* __restrict__ Wit, const uint32_t* __restrict__ bsrc,
         float* __restrict__ res) {
  __shared__ __align__(16) _Float16 bpack[12288];      // Bh [0..6143], Bp [6144..]
  __shared__ __align__(16) float it_f32[128];
  __shared__ __align__(16) _Float16 htile[4][16][128];
  __shared__ __align__(16) _Float16 ptile[4][16][128];
  __shared__ float rowterm[NPAD];
  __shared__ float w2s[NPAD];
  __shared__ float maskLDS[4][16];
  __shared__ float attLDS[4][16];
  __shared__ float curLDS[128];
  __shared__ float red[128];
  __shared__ float itn_s, b2s_s;

  const int bb = blockIdx.x;
  const int t  = threadIdx.x;

  // ---- Phase A: query vector, norm, per-row term, weight-pack copy --------
  {
    float v = (t < 64) ? item_emb[(size_t)item[bb] * Hh + t]
                       : time_emb[(size_t)timei[bb] * Hh + (t - 64)];
    it_f32[t] = v;
    red[t]    = v * v;
    curLDS[t] = 0.f;
  }
  for (int i = t; i < 6144; i += 128)
    ((uint32_t*)bpack)[i] = bsrc[i];
  __syncthreads();
  if (t == 0) {
    float s = 0.f;
    for (int i = 0; i < 128; ++i) s += red[i];
    itn_s = sqrtf(s);
    b2s_s = au_b2[0];
  }
  if (t < NPAD) {
    float s = (t < 36) ? au_b1[t] : -1e30f;      // pad -> relu kills it
    if (t < 36) {
      const float* w = Wit + t;
      for (int k = 0; k < 128; ++k) s += it_f32[k] * w[k * 36];
    }
    rowterm[t] = s;
    w2s[t] = (t < 36) ? au_w2[t] : 0.f;
  }
  __syncthreads();

  // ---- Phase B: per-wave history tiles ------------------------------------
  const int wv   = t >> 5;
  const int lane = t & 31;
  const int half = lane >> 4;           // 0: k 0..63 (item part), 1: 64..127 (time)
  const int r    = lane & 15;           // row in tile / N column
  const float itn = itn_s;

  float curacc[4] = {0.f, 0.f, 0.f, 0.f};

  for (int it8 = 0; it8 < 4; ++it8) {
    const int tile = wv + it8 * 4;      // wave-uniform
    if (tile >= NTILES_L) continue;

    // stage 16 history rows -> f16 h and h*it, plus dot/norm partials
    const int l = tile * 16 + r;
    const bool valid = (l < Ll);
    const float* ebase = nullptr;
    if (valid) {
      int iid = hist[((size_t)bb * Ll + l) * 3 + 0];
      int tid = hist[((size_t)bb * Ll + l) * 3 + 2];
      ebase = half ? (time_emb + (size_t)tid * Hh) : (item_emb + (size_t)iid * Hh);
    }
    _Float16* hrow = &htile[wv][r][half * 64];
    _Float16* prow = &ptile[wv][r][half * 64];
    const float* itp = &it_f32[half * 64];
    float dotv = 0.f, nrm = 0.f;
#pragma unroll 4
    for (int j = 0; j < 16; ++j) {
      float4 e  = valid ? ((const float4*)ebase)[j] : make_float4(0.f, 0.f, 0.f, 0.f);
      float4 iv = ((const float4*)itp)[j];
      dotv += e.x * iv.x + e.y * iv.y + e.z * iv.z + e.w * iv.w;
      nrm  += e.x * e.x + e.y * e.y + e.z * e.z + e.w * e.w;
      hrow[4 * j + 0] = (_Float16)e.x;  prow[4 * j + 0] = (_Float16)(e.x * iv.x);
      hrow[4 * j + 1] = (_Float16)e.y;  prow[4 * j + 1] = (_Float16)(e.y * iv.y);
      hrow[4 * j + 2] = (_Float16)e.z;  prow[4 * j + 2] = (_Float16)(e.z * iv.z);
      hrow[4 * j + 3] = (_Float16)e.w;  prow[4 * j + 3] = (_Float16)(e.w * iv.w);
    }
    dotv += __shfl_xor(dotv, 16);
    nrm  += __shfl_xor(nrm, 16);
    float sim = dotv / (sqrtf(nrm) * itn + EPSR);
    if (half == 0)
      maskLDS[wv][r] = (valid && sim >= THRE) ? 1.f : 0.f;
    wave_ds_sync();

    // WMMA: acc[nt] += h@Wh + (h*it)@Wp  over K=128
    v8f acc[3] = {};
#pragma unroll
    for (int kc = 0; kc < 4; ++kc) {
      Frag ah, ap;
      const _Float16* hb = &htile[wv][r][kc * 32 + half * 8];
      ah.q[0] = *(const uint4*)hb;
      ah.q[1] = *(const uint4*)(hb + 16);
      const _Float16* pb = &ptile[wv][r][kc * 32 + half * 8];
      ap.q[0] = *(const uint4*)pb;
      ap.q[1] = *(const uint4*)(pb + 16);
#pragma unroll
      for (int nt = 0; nt < 3; ++nt) {
        Frag bh, bp;
        const _Float16* bh0 = &bpack[((nt * 4 + kc) * 32 + lane) * 16];
        bh.q[0] = *(const uint4*)bh0;
        bh.q[1] = *(const uint4*)(bh0 + 8);
        bp.q[0] = *(const uint4*)(bh0 + 6144);
        bp.q[1] = *(const uint4*)(bh0 + 6144 + 8);
        acc[nt] = __builtin_amdgcn_wmma_f32_16x16x32_f16(
            false, ah.v, false, bh.v, (short)0, acc[nt], false, false);
        acc[nt] = __builtin_amdgcn_wmma_f32_16x16x32_f16(
            false, ap.v, false, bp.v, (short)0, acc[nt], false, false);
      }
    }

    // D layout: VGPR v -> M = v + 8*half, N column = lane&15 (+16*nt)
    float pv[8];
#pragma unroll
    for (int v = 0; v < 8; ++v) {
      float s = 0.f;
#pragma unroll
      for (int nt = 0; nt < 3; ++nt) {
        int n = nt * 16 + r;
        float z = acc[nt][v] + rowterm[n];
        s += fmaxf(z, 0.f) * w2s[n];
      }
      pv[v] = s;
    }
#pragma unroll
    for (int off = 1; off < 16; off <<= 1) {
#pragma unroll
      for (int v = 0; v < 8; ++v) pv[v] += __shfl_xor(pv[v], off);
    }
    if (r == 0) {
#pragma unroll
      for (int v = 0; v < 8; ++v) {
        int m = v + 8 * half;
        attLDS[wv][m] = maskLDS[wv][m] * (pv[v] + b2s_s);
      }
    }
    wave_ds_sync();

    // cur += att[m] * h[m][k]
#pragma unroll
    for (int j = 0; j < 4; ++j) {
      int k = lane + 32 * j;
      float s = 0.f;
#pragma unroll
      for (int m = 0; m < 16; ++m)
        s += attLDS[wv][m] * (float)htile[wv][m][k];
      curacc[j] += s;
    }
  }

#pragma unroll
  for (int j = 0; j < 4; ++j)
    atomicAdd(&curLDS[lane + 32 * j], curacc[j]);
  __syncthreads();

  // res = [u, it, c, cur]  (6H = 384)
  for (int i = t; i < 384; i += 128) {
    float vv;
    if (i < 64)        vv = user_emb[(size_t)user[bb] * Hh + i];
    else if (i < 192)  vv = it_f32[i - 64];
    else if (i < 256)  vv = cate_emb[(size_t)cate[bb] * Hh + (i - 192)];
    else               vv = curLDS[i - 256];
    res[(size_t)bb * 384 + i] = vv;
  }
}

// ---------------------------------------------------------------------------
// Final MLP with Dice (needs batch stats -> split into stages).
// st layout: [0:80] sum1 [80:160] sumsq1 [160:240] mean1 [240:320] rstd1
//            [320:360] sum2 [360:400] sumsq2 [400:440] mean2 [440:480] rstd2
// ---------------------------------------------------------------------------
__global__ void din_mlp1(const float* __restrict__ res,
                         const float* __restrict__ l1_w, const float* __restrict__ l1_b,
                         float* __restrict__ x1, float* __restrict__ st) {
  int idx = blockIdx.x * blockDim.x + threadIdx.x;
  if (idx >= Bb * 80) return;
  int b = idx / 80, j = idx % 80;
  const float* rr = res + (size_t)b * 384;
  float s = l1_b[j];
  for (int k = 0; k < 384; ++k) s += rr[k] * l1_w[k * 80 + j];
  x1[idx] = s;
  atomicAdd(&st[j], s);
  atomicAdd(&st[80 + j], s * s);
}

__global__ void din_stats(float* st, int sumoff, int meanoff, int n) {
  int j = threadIdx.x;
  if (j >= n) return;
  float mean = st[sumoff + j] * (1.0f / Bb);
  float var  = st[sumoff + n + j] * (1.0f / Bb) - mean * mean;
  st[meanoff + j]     = mean;
  st[meanoff + n + j] = rsqrtf(var + EPSR);
}

__global__ void din_mlp2(const float* __restrict__ x1,
                         const float* __restrict__ alpha1,
                         const float* __restrict__ l2_w, const float* __restrict__ l2_b,
                         float* __restrict__ x2, float* __restrict__ st) {
  int idx = blockIdx.x * blockDim.x + threadIdx.x;
  if (idx >= Bb * 40) return;
  int b = idx / 40, j = idx % 40;
  const float* xr = x1 + (size_t)b * 80;
  float s = l2_b[j];
  for (int k = 0; k < 80; ++k) {
    float x = xr[k];
    float z = (x - st[160 + k]) * st[240 + k];
    float p = 1.f / (1.f + __expf(-z));
    float xd = p * x + (1.f - p) * alpha1[k] * x;
    s += xd * l2_w[k * 40 + j];
  }
  x2[idx] = s;
  atomicAdd(&st[320 + j], s);
  atomicAdd(&st[360 + j], s * s);
}

__global__ void din_mlp3(const float* __restrict__ x2,
                         const float* __restrict__ alpha2,
                         const float* __restrict__ l3_w, const float* __restrict__ l3_b,
                         float* __restrict__ out, const float* __restrict__ st) {
  int idx = blockIdx.x * blockDim.x + threadIdx.x;
  if (idx >= Bb * 2) return;
  int b = idx / 2, o = idx % 2;
  const float* xr = x2 + (size_t)b * 40;
  float s = l3_b[o];
  for (int k = 0; k < 40; ++k) {
    float x = xr[k];
    float z = (x - st[400 + k]) * st[440 + k];
    float p = 1.f / (1.f + __expf(-z));
    float xd = p * x + (1.f - p) * alpha2[k] * x;
    s += xd * l3_w[k * 2 + o];
  }
  out[idx] = s;
}

// ---------------------------------------------------------------------------
extern "C" void kernel_launch(void* const* d_in, const int* in_sizes, int n_in,
                              void* d_out, int out_size, void* d_ws, size_t ws_size,
                              hipStream_t stream) {
  const int*   user     = (const int*)d_in[0];
  const int*   item     = (const int*)d_in[1];
  const int*   cate     = (const int*)d_in[2];
  const int*   timei    = (const int*)d_in[3];
  const int*   hist     = (const int*)d_in[4];
  const float* user_emb = (const float*)d_in[5];
  const float* item_emb = (const float*)d_in[6];
  const float* cate_emb = (const float*)d_in[7];
  const float* time_emb = (const float*)d_in[8];
  const float* au_w1    = (const float*)d_in[9];
  const float* au_b1    = (const float*)d_in[10];
  const float* au_w2    = (const float*)d_in[11];
  const float* au_b2    = (const float*)d_in[12];
  const float* l1_w     = (const float*)d_in[13];
  const float* l1_b     = (const float*)d_in[14];
  const float* alpha1   = (const float*)d_in[15];
  const float* l2_w     = (const float*)d_in[16];
  const float* l2_b     = (const float*)d_in[17];
  const float* alpha2   = (const float*)d_in[18];
  const float* l3_w     = (const float*)d_in[19];
  const float* l3_b     = (const float*)d_in[20];
  float* out = (float*)d_out;

  char* ws = (char*)d_ws;
  // workspace layout (≈8.3 MB total)
  _Float16* Bh  = (_Float16*)(ws + 0);          // 12288 B
  _Float16* Bp  = (_Float16*)(ws + 12288);      // 12288 B
  float*    Wit = (float*)(ws + 24576);         // 18432 B
  float*    st  = (float*)(ws + 43008);         // 1920 B
  float*    res = (float*)(ws + 45056);         // 6291456 B
  float*    x1  = (float*)(ws + 45056 + 6291456);             // 1310720 B
  float*    x2  = (float*)(ws + 45056 + 6291456 + 1310720);   // 655360 B

  hipMemsetAsync(st, 0, 480 * sizeof(float), stream);

  din_prep<<<24, 256, 0, stream>>>(au_w1, Wit, Bh, Bp);

  din_attn<<<Bb, 128, 0, stream>>>(user, item, cate, timei, hist,
                                   user_emb, item_emb, cate_emb, time_emb,
                                   au_b1, au_w2, au_b2,
                                   Wit, (const uint32_t*)Bh, res);

  din_mlp1<<<(Bb * 80 + 255) / 256, 256, 0, stream>>>(res, l1_w, l1_b, x1, st);
  din_stats<<<1, 128, 0, stream>>>(st, 0, 160, 80);
  din_mlp2<<<(Bb * 40 + 255) / 256, 256, 0, stream>>>(x1, alpha1, l2_w, l2_b, x2, st);
  din_stats<<<1, 128, 0, stream>>>(st, 320, 400, 40);
  din_mlp3<<<(Bb * 2 + 255) / 256, 256, 0, stream>>>(x2, alpha2, l3_w, l3_b, out, st);
}
